// GCN_2199023255969
// MI455X (gfx1250) — matrix operand order repaired and checked by reading, not compiled, not run on hardware
//
#include <hip/hip_runtime.h>
#include <hip/hip_bf16.h>

// ---------------------------------------------------------------------------
// GCN forward on MI455X (gfx1250, wave32, WMMA).
//   h1 = relu(spmm(x@W1) + b1)
//   h2 = relu(spmm(h1@Wh) + bh)
//   out = log_softmax(spmm(h2@W2) + b2)
// GEMMs: v_wmma_f32_16x16x32_bf16. A is bf16 row-major (tile = two b128
// loads/lane), B is pre-swizzled bf16 in the WMMA lane layout (tile = two
// contiguous b128 loads/lane). Each wave owns TWO 16-row A tiles so every B
// tile feeds 2 WMMAs (B-reuse x2, more independent WMMA work per loadcnt
// wait). SpMM: feature-parallel edges + native f32 atomics (operands are
// L2-resident: 25.6 MB << 192 MB L2).
// ---------------------------------------------------------------------------

typedef __attribute__((ext_vector_type(16))) __bf16 v16bf;
typedef __attribute__((ext_vector_type(8)))  float  v8f;

union Tile16 {            // 16 bf16 = 8 VGPRs = 32 bytes
    v16bf v;
    uint4 q[2];
};

union PackBF2 { __bf16 h[2]; unsigned u; };

// ---------------------------------------------------------------------------
// Dense GEMM:  C[nrows x ND] = A[nrows x KD] @ Wswz
// MT row tiles (16 rows each) per wave, NT = ND/16 col tiles, K steps of 32.
// ---------------------------------------------------------------------------
template <int KD, int ND, int MT>
__global__ void gcn_gemm_wmma(const __bf16* __restrict__ A,
                              const __bf16* __restrict__ Wswz,
                              float* __restrict__ C, int nrows) {
    const int lane = threadIdx.x & 31;
    const int wave = threadIdx.x >> 5;
    const int row0 = (blockIdx.x * (blockDim.x >> 5) + wave) * (16 * MT);
    if (row0 >= nrows) return;                 // uniform per wave: EXEC stays all-1s

    const int m  = lane & 15;
    const int hi = lane >> 4;

    constexpr int NT = ND / 16;
    v8f acc[MT][NT];
#pragma unroll
    for (int i = 0; i < MT; ++i)
#pragma unroll
        for (int t = 0; t < NT; ++t)
            acc[i][t] = (v8f){0.f, 0.f, 0.f, 0.f, 0.f, 0.f, 0.f, 0.f};

    const uint4* Arow[MT];
#pragma unroll
    for (int i = 0; i < MT; ++i) {
        const int r = row0 + i * 16 + m;
        const int rc = (r < nrows) ? r : (nrows - 1);
        Arow[i] = (const uint4*)(A + (size_t)rc * KD);   // 8 halfs per uint4
    }
    const uint4* Bs = (const uint4*)Wswz;

#pragma unroll
    for (int kb = 0; kb < KD / 32; ++kb) {
        // A tiles (16x32 bf16): lane needs K=[hi*16..+15]
        Tile16 a[MT];
#pragma unroll
        for (int i = 0; i < MT; ++i) {
            a[i].q[0] = Arow[i][kb * 4 + 2 * hi];
            a[i].q[1] = Arow[i][kb * 4 + 2 * hi + 1];
        }
#pragma unroll
        for (int t = 0; t < NT; ++t) {
            Tile16 b;
            const size_t base = ((size_t)(kb * NT + t) * 32 + lane) * 2;
            b.q[0] = Bs[base];
            b.q[1] = Bs[base + 1];
#pragma unroll
            for (int i = 0; i < MT; ++i)
                acc[i][t] = __builtin_amdgcn_wmma_f32_16x16x32_bf16(
                    false, a[i].v, false, b.v, (short)0, acc[i][t], false, false);
        }
    }

#pragma unroll
    for (int i = 0; i < MT; ++i) {
#pragma unroll
        for (int t = 0; t < NT; ++t) {
            const int col = t * 16 + (lane & 15);
#pragma unroll
            for (int r = 0; r < 8; ++r) {
                const int row = row0 + i * 16 + (hi << 3) + r;
                if (row < nrows) C[(size_t)row * ND + col] = acc[i][t][r];
            }
        }
    }
}

// ---------------------------------------------------------------------------
// Swizzle W[KD x ND] (f32 row-major) -> bf16 WMMA B lane layout:
// out[((kb*NT+t)*32 + lane)*16 + j] = bf16(W[(kb*32 + (lane>>4)*16 + j)*ND + t*16 + (lane&15)])
// ---------------------------------------------------------------------------
template <int KD, int ND>
__global__ void gcn_swizzle_w(const float* __restrict__ W, __bf16* __restrict__ out) {
    const int gid = blockIdx.x * blockDim.x + threadIdx.x;
    if (gid >= KD * ND) return;
    constexpr int NT = ND / 16;
    const int j    = gid & 15;
    const int lane = (gid >> 4) & 31;
    const int blk  = gid >> 9;
    const int t    = blk % NT;
    const int kb   = blk / NT;
    const int k = kb * 32 + (lane >> 4) * 16 + j;
    const int n = t * 16 + (lane & 15);
    out[gid] = (__bf16)W[(size_t)k * ND + n];
}

// ---------------------------------------------------------------------------
// Streaming f32 -> bf16 (2 elems/thread, packed 4-byte store).
// ---------------------------------------------------------------------------
__global__ void gcn_f32_to_bf16(const float* __restrict__ in,
                                unsigned* __restrict__ out, int total2) {
    const int gid = blockIdx.x * blockDim.x + threadIdx.x;
    if (gid >= total2) return;
    const float2 f = ((const float2*)in)[gid];
    PackBF2 p;
    p.h[0] = (__bf16)f.x;
    p.h[1] = (__bf16)f.y;
    out[gid] = p.u;
}

// hb = bf16(relu(h)), 2 elems/thread
__global__ void gcn_relu_bf16(const float* __restrict__ h,
                              unsigned* __restrict__ hb, int total2) {
    const int gid = blockIdx.x * blockDim.x + threadIdx.x;
    if (gid >= total2) return;
    const float2 f = ((const float2*)h)[gid];
    PackBF2 p;
    p.h[0] = (__bf16)fmaxf(f.x, 0.0f);
    p.h[1] = (__bf16)fmaxf(f.y, 0.0f);
    hb[gid] = p.u;
}

// ---------------------------------------------------------------------------
// out[i] = bias[i & (D-1)]
// ---------------------------------------------------------------------------
__global__ void gcn_bias_init(float* __restrict__ out, const float* __restrict__ b,
                              int total, int dmask) {
    const int gid = blockIdx.x * blockDim.x + threadIdx.x;
    if (gid < total) out[gid] = b[gid & dmask];
}

// ---------------------------------------------------------------------------
// SpMM scatter: out[dst[e]*D + f] += edge_w[e] * sup[src[e]*D + f]
// D consecutive lanes per edge -> coalesced gather + dense atomic addresses.
// ---------------------------------------------------------------------------
template <int LOG2D>
__global__ void gcn_spmm_atomic(const int* __restrict__ src, const int* __restrict__ dst,
                                const float* __restrict__ ew,
                                const float* __restrict__ sup,
                                float* __restrict__ out, int total) {
    const int gid = blockIdx.x * blockDim.x + threadIdx.x;
    if (gid >= total) return;
    constexpr int D = 1 << LOG2D;
    const int e = gid >> LOG2D;
    const int f = gid & (D - 1);
    const int s = src[e];
    const int d = dst[e];
    const float w = ew[e];
    unsafeAtomicAdd(&out[(size_t)d * D + f], w * sup[(size_t)s * D + f]);
}

// ---------------------------------------------------------------------------
// Row-wise log_softmax over 16 classes.
// ---------------------------------------------------------------------------
__global__ void gcn_log_softmax16(const float* __restrict__ in,
                                  float* __restrict__ out, int nrows) {
    const int r = blockIdx.x * blockDim.x + threadIdx.x;
    if (r >= nrows) return;
    const float* p = in + (size_t)r * 16;
    float m = p[0];
#pragma unroll
    for (int i = 1; i < 16; ++i) m = fmaxf(m, p[i]);
    float s = 0.0f;
#pragma unroll
    for (int i = 0; i < 16; ++i) s += __expf(p[i] - m);
    const float l = __logf(s);
    float* q = out + (size_t)r * 16;
#pragma unroll
    for (int i = 0; i < 16; ++i) q[i] = p[i] - m - l;
}

// ---------------------------------------------------------------------------
extern "C" void kernel_launch(void* const* d_in, const int* in_sizes, int n_in,
                              void* d_out, int out_size, void* d_ws, size_t ws_size,
                              hipStream_t stream) {
    const float* x   = (const float*)d_in[0];   // [N,256]
    const int*   src = (const int*)  d_in[1];   // [E]
    const int*   dst = (const int*)  d_in[2];   // [E]
    const float* ew  = (const float*)d_in[3];   // [E]
    const float* W1  = (const float*)d_in[4];   // [256,64]
    const float* b1  = (const float*)d_in[5];   // [64]
    const float* Wh  = (const float*)d_in[6];   // [64,64]
    const float* bh  = (const float*)d_in[7];   // [64]
    const float* W2  = (const float*)d_in[8];   // [64,16]
    const float* b2  = (const float*)d_in[9];   // [16]
    float* out = (float*)d_out;

    const int N = in_sizes[0] / 256;            // 100000
    const int E = in_sizes[1];                  // 1600000

    // Workspace arena (~115.3 MB):
    //   sup f32[N*64] | h f32[N*64] | hb bf16[N*64] | xb bf16[N*256] | swizzled weights
    float*  sup = (float*)d_ws;
    float*  h   = sup + (size_t)N * 64;
    __bf16* hb  = (__bf16*)(h + (size_t)N * 64);
    __bf16* xb  = hb + (size_t)N * 64;
    __bf16* w1s = xb + (size_t)N * 256;
    __bf16* whs = w1s + 256 * 64;
    __bf16* w2s = whs + 64 * 64;

    const dim3 blk(256);
    const int gemmGrid = (N + 255) / 256;       // 8 waves/block * 32 rows/wave
    const int nd64 = N * 64, nd16 = N * 16;
    const int ed64 = E * 64, ed16 = E * 16;
    const int x2 = N * 128;                     // (N*256)/2 packed pairs
    const int h2 = N * 32;                      // (N*64)/2 packed pairs

    // ---- one-time format prep (part of the deterministic pipeline) ----
    gcn_f32_to_bf16<<<(x2 + 255) / 256, blk, 0, stream>>>(x, (unsigned*)xb, x2);
    gcn_swizzle_w<256, 64><<<(256 * 64 + 255) / 256, blk, 0, stream>>>(W1, w1s);
    gcn_swizzle_w<64, 64><<<(64 * 64 + 255) / 256, blk, 0, stream>>>(Wh, whs);
    gcn_swizzle_w<64, 16><<<(64 * 16 + 255) / 256, blk, 0, stream>>>(W2, w2s);

    // ---- layer 1 ----
    gcn_gemm_wmma<256, 64, 2><<<gemmGrid, blk, 0, stream>>>(xb, w1s, sup, N);
    gcn_bias_init<<<(nd64 + 255) / 256, blk, 0, stream>>>(h, b1, nd64, 63);
    gcn_spmm_atomic<6><<<(ed64 + 255) / 256, blk, 0, stream>>>(src, dst, ew, sup, h, ed64);
    gcn_relu_bf16<<<(h2 + 255) / 256, blk, 0, stream>>>(h, (unsigned*)hb, h2);

    // ---- layer 2 ----
    gcn_gemm_wmma<64, 64, 2><<<gemmGrid, blk, 0, stream>>>(hb, whs, sup, N);
    gcn_bias_init<<<(nd64 + 255) / 256, blk, 0, stream>>>(h, bh, nd64, 63);
    gcn_spmm_atomic<6><<<(ed64 + 255) / 256, blk, 0, stream>>>(src, dst, ew, sup, h, ed64);
    gcn_relu_bf16<<<(h2 + 255) / 256, blk, 0, stream>>>(h, (unsigned*)hb, h2);

    // ---- layer 3 ----
    gcn_gemm_wmma<64, 16, 2><<<gemmGrid, blk, 0, stream>>>(hb, w2s, sup, N);
    gcn_bias_init<<<(nd16 + 255) / 256, blk, 0, stream>>>(h, b2, nd16, 15);
    gcn_spmm_atomic<4><<<(ed16 + 255) / 256, blk, 0, stream>>>(src, dst, ew, sup, h, ed16);
    gcn_log_softmax16<<<(N + 255) / 256, blk, 0, stream>>>(h, out, N);
}
